// RNNTModel_87213605912902
// MI455X (gfx1250) — matrix operand order repaired and checked by reading, not compiled, not run on hardware
//
#include <hip/hip_runtime.h>

#define B_ 4
#define T_ 256
#define U_ 128
#define D_ 512
#define C_ 1024

typedef __attribute__((ext_vector_type(16))) __bf16 v16bf;
typedef __attribute__((ext_vector_type(8)))  float  v8f;

// round-to-nearest-even f32 -> bf16 (returns low 16 bits)
__device__ __forceinline__ unsigned f2bf(float f) {
    unsigned u = __builtin_bit_cast(unsigned, f);
    return (u + 0x7FFFu + ((u >> 16) & 1u)) >> 16;
}

__device__ __forceinline__ float fast_tanh(float x) {
    x = fminf(fmaxf(x, -15.f), 15.f);
#if __has_builtin(__builtin_amdgcn_tanhf)
    return __builtin_amdgcn_tanhf(x);   // gfx1250 v_tanh_f32 (TRANS unit)
#else
    float e = __expf(-2.f * x);
    return (1.f - e) / (1.f + e);
#endif
}

// ---------------------------------------------------------------------------
// Stage 1: enc_proj = enc @ W_e + b1 ; dec_proj = dec @ W_d   (f32, small)
// ---------------------------------------------------------------------------
__global__ __launch_bounds__(256) void proj4(
    const float* __restrict__ enc, const float* __restrict__ dec,
    const float* __restrict__ We,  const float* __restrict__ Wd,
    const float* __restrict__ b1,
    float* __restrict__ encp, float* __restrict__ decp)
{
    __shared__ float xs[4][D_];
    const int blk = blockIdx.x;
    const int nEnc = (B_ * T_) / 4;
    const float* x; const float* W; float* o; bool useb;
    if (blk < nEnc) { int r0 = blk * 4;          x = enc + (size_t)r0 * D_; W = We; o = encp + (size_t)r0 * D_; useb = true;  }
    else            { int r0 = (blk - nEnc) * 4; x = dec + (size_t)r0 * D_; W = Wd; o = decp + (size_t)r0 * D_; useb = false; }

    for (int i = threadIdx.x; i < 4 * D_; i += 256) xs[i >> 9][i & (D_ - 1)] = x[i];
    __syncthreads();

    const int h0 = threadIdx.x;
    float a[4][2];
    #pragma unroll
    for (int j = 0; j < 4; ++j) {
        a[j][0] = useb ? b1[h0]       : 0.f;
        a[j][1] = useb ? b1[h0 + 256] : 0.f;
    }
    for (int k = 0; k < D_; ++k) {
        float w0 = W[(size_t)k * D_ + h0];
        float w1 = W[(size_t)k * D_ + h0 + 256];
        #pragma unroll
        for (int j = 0; j < 4; ++j) {
            a[j][0] = fmaf(xs[j][k], w0, a[j][0]);
            a[j][1] = fmaf(xs[j][k], w1, a[j][1]);
        }
    }
    #pragma unroll
    for (int j = 0; j < 4; ++j) {
        o[(size_t)j * D_ + h0]       = a[j][0];
        o[(size_t)j * D_ + h0 + 256] = a[j][1];
    }
}

// ---------------------------------------------------------------------------
// Stage 2: pack W2 (D x NCLS, f32 row-major) into bf16 B-fragment order.
// dword index = ((kt*64 + nt)*32 + lane)*8 + v
// B packing used here (k = 16h+s): VGPR v, lanes 0-15: N=lane,    K=2v,2v+1
//                                   lanes16-31: N=lane-16, K=16+2v,17+2v
// ---------------------------------------------------------------------------
__global__ __launch_bounds__(256) void pack_w2(
    const float* __restrict__ W2, unsigned* __restrict__ w2p)
{
    const int tid  = blockIdx.x * 256 + threadIdx.x;   // 0 .. 262143
    const int v    = tid & 7;
    const int lane = (tid >> 3) & 31;
    const int nt   = (tid >> 8) & 63;
    const int kt   = tid >> 14;
    const int k = kt * 32 + 2 * v + ((lane & 16) ? 16 : 0);
    const int n = nt * 16 + (lane & 15);
    unsigned lo = f2bf(W2[(size_t)k * C_ + n]);
    unsigned hi = f2bf(W2[(size_t)(k + 1) * C_ + n]);
    w2p[tid] = lo | (hi << 16);
}

// ---------------------------------------------------------------------------
// Stage 3: fused joint. Grid (U/32, T, B); 256 threads = 8 waves.
// WG covers 32 u-rows x all 1024 classes; wave w owns N in [128w, 128w+128).
// Each wave: 2 M-tiles x 8 N-tiles (128 f32 acc VGPRs); B frag reused 2x.
// hsum: 32x512 f32 (64KB) with XOR bank swizzle on addr bits [5:2].
// ---------------------------------------------------------------------------
__global__ __launch_bounds__(256) void rnnt_joint(
    const float* __restrict__ encp, const float* __restrict__ decp,
    const unsigned* __restrict__ w2p, float* __restrict__ out)
{
    __shared__ float hsum[32 * D_];          // 64KB; aliased as softmax scratch later

    const int ub = blockIdx.x, t = blockIdx.y, b = blockIdx.z;
    const int u0 = ub * 32;
    const float* er = encp + (size_t)(b * T_ + t) * D_;
    const float* dr = decp + (size_t)(b * U_ + u0) * D_;

    for (int i = threadIdx.x; i < 32 * D_; i += 256) {
        int m = i >> 9, k = i & (D_ - 1);
        hsum[m * D_ + (k ^ ((m & 15) << 2))] = dr[i] + er[k];  // b1 folded into encp
    }
    __syncthreads();

    const int lane = threadIdx.x & 31, wave = threadIdx.x >> 5;
    const int half = lane >> 4, nlo = lane & 15;
    const int swz  = nlo << 2;               // XOR swizzle for this lane's rows

    v8f acc[2][8] = {};

    #pragma unroll 1
    for (int kt = 0; kt < 16; ++kt) {
        // A fragments (one per M-tile): contiguous k order, same as packed B.
        // K(p) = 2p + (lane>=16?16:0)
        union { v16bf v; unsigned u[8]; } A[2];
        const int kb = kt * 32 + half * 16;
        #pragma unroll
        for (int mt = 0; mt < 2; ++mt) {
            const float* row = &hsum[(mt * 16 + nlo) * D_];
            #pragma unroll
            for (int p = 0; p < 8; ++p) {
                int k  = kb + 2 * p;
                int ks = k ^ swz;            // k even; swizzle touches bits[5:2]
                float x0 = fast_tanh(row[ks]);
                float x1 = fast_tanh(row[ks + 1]);
                A[mt].u[p] = f2bf(x0) | (f2bf(x1) << 16);
            }
        }
        const unsigned* wp = w2p + (((size_t)kt * 64 + wave * 8) * 32 + lane) * 8;
        #pragma unroll
        for (int tt = 0; tt < 8; ++tt) {
            union { v16bf v; uint4 q[2]; } Bf;
            const uint4* src = (const uint4*)(wp + tt * 256);
            Bf.q[0] = src[0];
            Bf.q[1] = src[1];
            acc[0][tt] = __builtin_amdgcn_wmma_f32_16x16x32_bf16(
                false, A[0].v, false, Bf.v, (short)0, acc[0][tt], false, false);
            acc[1][tt] = __builtin_amdgcn_wmma_f32_16x16x32_bf16(
                false, A[1].v, false, Bf.v, (short)0, acc[1][tt], false, false);
        }
    }

    __syncthreads();                 // hsum dead; reuse as cross-wave scratch
    float* red = hsum;               // [phase(2)][wave(8)][row(32)]

    // ---- log_softmax over NCLS=1024. Row (mt*16 + half*8 + r) lives in VGPR r
    // of one 16-lane half; reduce within half via shfl_xor, then across waves.
    float rmax[2][8], lz[2][8];
    #pragma unroll
    for (int mt = 0; mt < 2; ++mt)
    #pragma unroll
    for (int r = 0; r < 8; ++r) {
        float m = acc[mt][0][r];
        #pragma unroll
        for (int tt = 1; tt < 8; ++tt) m = fmaxf(m, acc[mt][tt][r]);
        #pragma unroll
        for (int s = 1; s < 16; s <<= 1) m = fmaxf(m, __shfl_xor(m, s, 32));
        rmax[mt][r] = m;
    }
    if (nlo == 0) {
        #pragma unroll
        for (int mt = 0; mt < 2; ++mt)
        #pragma unroll
        for (int r = 0; r < 8; ++r)
            red[wave * 32 + mt * 16 + half * 8 + r] = rmax[mt][r];
    }
    __syncthreads();
    #pragma unroll
    for (int mt = 0; mt < 2; ++mt)
    #pragma unroll
    for (int r = 0; r < 8; ++r) {
        int idx = mt * 16 + half * 8 + r;
        float m = red[idx];
        #pragma unroll
        for (int w = 1; w < 8; ++w) m = fmaxf(m, red[w * 32 + idx]);
        rmax[mt][r] = m;
    }
    #pragma unroll
    for (int mt = 0; mt < 2; ++mt)
    #pragma unroll
    for (int r = 0; r < 8; ++r) {
        float s = 0.f;
        #pragma unroll
        for (int tt = 0; tt < 8; ++tt) s += __expf(acc[mt][tt][r] - rmax[mt][r]);
        #pragma unroll
        for (int k = 1; k < 16; k <<= 1) s += __shfl_xor(s, k, 32);
        lz[mt][r] = s;
    }
    if (nlo == 0) {                  // phase-1 region disjoint from phase-0
        #pragma unroll
        for (int mt = 0; mt < 2; ++mt)
        #pragma unroll
        for (int r = 0; r < 8; ++r)
            red[256 + wave * 32 + mt * 16 + half * 8 + r] = lz[mt][r];
    }
    __syncthreads();
    #pragma unroll
    for (int mt = 0; mt < 2; ++mt)
    #pragma unroll
    for (int r = 0; r < 8; ++r) {
        int idx = mt * 16 + half * 8 + r;
        float s = red[256 + idx];
        #pragma unroll
        for (int w = 1; w < 8; ++w) s += red[256 + w * 32 + idx];
        lz[mt][r] = rmax[mt][r] + __logf(s);
    }

    // ---- store: out[((b*T+t)*U + u0 + row)*C + n]
    float* ob = out + ((size_t)(b * T_ + t) * U_ + u0) * C_;
    #pragma unroll
    for (int tt = 0; tt < 8; ++tt) {
        int n = wave * 128 + tt * 16 + nlo;
        #pragma unroll
        for (int mt = 0; mt < 2; ++mt)
        #pragma unroll
        for (int r = 0; r < 8; ++r) {
            int u = mt * 16 + half * 8 + r;
            ob[(size_t)u * C_ + n] = acc[mt][tt][r] - lz[mt][r];
        }
    }
}

// ---------------------------------------------------------------------------
extern "C" void kernel_launch(void* const* d_in, const int* in_sizes, int n_in,
                              void* d_out, int out_size, void* d_ws, size_t ws_size,
                              hipStream_t stream)
{
    const float* enc = (const float*)d_in[0];
    const float* dec = (const float*)d_in[1];
    const float* We  = (const float*)d_in[2];
    const float* Wd  = (const float*)d_in[3];
    const float* b1  = (const float*)d_in[4];
    const float* W2  = (const float*)d_in[5];

    float*    encp = (float*)d_ws;                             // 2MB
    float*    decp = encp + (size_t)B_ * T_ * D_;              // 1MB
    unsigned* w2p  = (unsigned*)(decp + (size_t)B_ * U_ * D_); // 1MB (bf16 packed)

    proj4  <<<(B_ * T_ + B_ * U_) / 4, 256, 0, stream>>>(enc, dec, We, Wd, b1, encp, decp);
    pack_w2<<<(D_ * C_ / 2) / 256,     256, 0, stream>>>(W2, w2p);
    rnnt_joint<<<dim3(U_ / 32, T_, B_), 256, 0, stream>>>(encp, decp, w2p, (float*)d_out);
}
